// LIIF3D_51049981280699
// MI455X (gfx1250) — matrix operand order repaired and checked by reading, not compiled, not run on hardware
//
#include <hip/hip_runtime.h>

typedef __attribute__((ext_vector_type(16))) _Float16 v16h;
typedef __attribute__((ext_vector_type(8)))  _Float16 v8h;
typedef __attribute__((ext_vector_type(8)))  float    v8f;

#define QTOT 131072
#define HID  256
#define KP0  96     // 71 input features padded to 96 (3 K-steps of 32)

// ---------------------------------------------------------------------------
// Preamble 1: volume (1,64,32,32,32) f32 channel-major  ->  f16 channels-last
// vol2[v][c], v = z*1024 + y*32 + x.  2,097,152 elements.
// ---------------------------------------------------------------------------
__global__ void liif_conv_vol(const float* __restrict__ inp,
                              _Float16* __restrict__ vol2) {
  int idx = blockIdx.x * blockDim.x + threadIdx.x;   // exact grid, no bounds
  int v = idx >> 6, c = idx & 63;
  vol2[idx] = (_Float16)inp[c * 32768 + v];
}

// ---------------------------------------------------------------------------
// Preamble 2: weights f32 -> f16 row-major [K][256] (B-fragment friendly).
// W0 zero-padded from 71 to 96 rows.
// ---------------------------------------------------------------------------
__global__ void liif_conv_w(const float* __restrict__ W0,
                            const float* __restrict__ W1,
                            const float* __restrict__ W2,
                            _Float16* __restrict__ W0h,
                            _Float16* __restrict__ W1h,
                            _Float16* __restrict__ W2h) {
  int idx = blockIdx.x * blockDim.x + threadIdx.x;   // 155648 total, exact
  if (idx < KP0 * HID) {
    int k = idx / HID, n = idx % HID;
    W0h[idx] = (k < 71) ? (_Float16)W0[k * HID + n] : (_Float16)0.0f;
  } else if (idx < KP0 * HID + HID * HID) {
    int j = idx - KP0 * HID;
    W1h[j] = (_Float16)W1[j];
  } else {
    int j = idx - KP0 * HID - HID * HID;
    W2h[j] = (_Float16)W2[j];
  }
}

// ---------------------------------------------------------------------------
// Per-shift geometry, faithful to the reference (incl. its axis ordering in
// rel_coord: channel0 of the coord volume varies along W/x, etc.)
// ---------------------------------------------------------------------------
__device__ __forceinline__ float lgrid(int i) {
  return -1.0f + (2.0f / 31.0f) * (float)i;   // linspace(-1,1,32)
}

struct Geo {
  int z0, z1, y0, y1, x0, x1;
  float wz, wy, wx;
  float rel0, rel1, rel2, area;
};

__device__ __forceinline__ Geo geom(float cd, float ch, float cw, int s) {
  const float rr = 1.0f / 32.0f;       // rx = ry = rz = 2/D/2
  const float EPSF = 1e-6f;
  float sd = ((s & 4) ? rr : -rr) + EPSF;   // vx outer loop
  float sh = ((s & 2) ? rr : -rr) + EPSF;   // vy
  float sw = ((s & 1) ? rr : -rr) + EPSF;   // vz inner loop
  float gd = fminf(fmaxf(cd + sd, -1.0f + 1e-6f), 1.0f - 1e-6f);
  float gh = fminf(fmaxf(ch + sh, -1.0f + 1e-6f), 1.0f - 1e-6f);
  float gw = fminf(fmaxf(cw + sw, -1.0f + 1e-6f), 1.0f - 1e-6f);
  float z = (gd + 1.0f) * 16.0f - 0.5f;
  float y = (gh + 1.0f) * 16.0f - 0.5f;
  float x = (gw + 1.0f) * 16.0f - 0.5f;
  float zf = floorf(z), yf = floorf(y), xf = floorf(x);
  Geo g;
  g.wz = z - zf; g.wy = y - yf; g.wx = x - xf;
  int zi = (int)zf, yi = (int)yf, xi = (int)xf;
  g.z0 = min(max(zi, 0), 31);     g.z1 = min(max(zi + 1, 0), 31);
  g.y0 = min(max(yi, 0), 31);     g.y1 = min(max(yi + 1, 0), 31);
  g.x0 = min(max(xi, 0), 31);     g.x1 = min(max(xi + 1, 0), 31);
  // q_coord: channel0 = x-lerp, channel1 = y-lerp, channel2 = z-lerp
  float qx = (1.0f - g.wx) * lgrid(g.x0) + g.wx * lgrid(g.x1);
  float qy = (1.0f - g.wy) * lgrid(g.y0) + g.wy * lgrid(g.y1);
  float qz = (1.0f - g.wz) * lgrid(g.z0) + g.wz * lgrid(g.z1);
  g.rel0 = (cd - qx) * 32.0f;    // reference pairs coord dim0 with q_coord ch0
  g.rel1 = (ch - qy) * 32.0f;
  g.rel2 = (cw - qz) * 32.0f;
  g.area = fabsf(g.rel0 * g.rel1 * g.rel2) + 1e-9f;
  return g;
}

// ---------------------------------------------------------------------------
// One MLP layer: D(16x256) = relu( A(16x K) * W(K x 256) + b ), via
// v_wmma_f32_16x16x32_f16.  A fragments from LDS per ISA 16-bit A layout,
// B fragments = 16 halves of one f16 weight column, stride HID (lane = N).
// Epilogue writes relu'd f16 back to LDS in the C/D layout (lane = N,
// vgpr = M row).
// ---------------------------------------------------------------------------
template <int KSTEPS>
__device__ __forceinline__ void mlp_layer(const _Float16* inl, int instride,
                                          _Float16* outl,
                                          const _Float16* __restrict__ Wh,
                                          const float* __restrict__ bias,
                                          int lane) {
  v8f zero = {};
  v8f acc[16];
#pragma unroll
  for (int nt = 0; nt < 16; ++nt) acc[nt] = zero;
  const int row = lane & 15;
  const int sel = (lane < 16) ? 0 : 8;     // K sub-block per ISA A layout
#pragma unroll 1
  for (int ks = 0; ks < KSTEPS; ++ks) {
    const _Float16* ap = inl + row * instride + ks * 32 + sel;
    v8h lo = *(const v8h*)ap;              // K {0..7} / {8..15}
    v8h hi = *(const v8h*)(ap + 16);       // K {16..23} / {24..31}
    v16h a;
#pragma unroll
    for (int t = 0; t < 8; ++t) { a[t] = lo[t]; a[t + 8] = hi[t]; }
    const _Float16* bp = Wh + (ks * 32 + sel) * HID + row;   // B: lane = N, elems = K (table order)
#pragma unroll
    for (int nt = 0; nt < 16; ++nt) {
      v16h b;
#pragma unroll
      for (int t = 0; t < 8; ++t) {
        b[t] = bp[t * HID + nt * 16];
        b[t + 8] = bp[(t + 16) * HID + nt * 16];
      }
      acc[nt] = __builtin_amdgcn_wmma_f32_16x16x32_f16(
          false, a, false, b, (short)0, acc[nt], false, false);
    }
  }
  const int ncol = lane & 15;
  const int mbase = (lane < 16) ? 0 : 8;   // C/D layout
#pragma unroll
  for (int nt = 0; nt < 16; ++nt) {
    int n = nt * 16 + ncol;
    float bv = bias[n];
#pragma unroll
    for (int rr = 0; rr < 8; ++rr) {
      float v = fmaxf(acc[nt][rr] + bv, 0.0f);
      outl[(mbase + rr) * HID + n] = (_Float16)v;
    }
  }
}

// ---------------------------------------------------------------------------
// Main kernel: 2 waves / block, 16 queries / wave, 32 KB LDS (ping-pong).
// ---------------------------------------------------------------------------
__global__ void __launch_bounds__(64)
liif_main(const _Float16* __restrict__ vol2,
          const float* __restrict__ coord, const float* __restrict__ cell,
          const _Float16* __restrict__ W0h, const float* __restrict__ b0,
          const _Float16* __restrict__ W1h, const float* __restrict__ b1,
          const _Float16* __restrict__ W2h, const float* __restrict__ b2,
          const float* __restrict__ W3, const float* __restrict__ b3,
          float* __restrict__ out) {
  __shared__ __align__(16) _Float16 lds[2][2][16 * HID];
  const int wave = threadIdx.x >> 5;
  const int lane = threadIdx.x & 31;
  const int r = lane & 15;                       // query within wave tile
  const int q = (blockIdx.x * 2 + wave) * 16 + r;

  const float cd  = coord[q * 3 + 0];
  const float chc = coord[q * 3 + 1];
  const float cw  = coord[q * 3 + 2];
  const float cl0 = cell[q * 3 + 0];
  const float cl1 = cell[q * 3 + 1];
  const float cl2 = cell[q * 3 + 2];
  const float rc0 = cl0 * 32.0f, rc1 = cl1 * 32.0f, rc2 = cl2 * 32.0f;
  const float cv  = fabsf(cl0 * cl1 * cl2);
  const float ssq = fminf(fmaxf(8.0f / (cv + 1e-8f), 1.0f), 64.0f);

  float tot = 0.0f;
#pragma unroll
  for (int s = 0; s < 8; ++s) tot += geom(cd, chc, cw, s).area;

  float outacc = 0.0f;
  _Float16* bufA = &lds[wave][0][0];   // H0 / H2
  _Float16* bufB = &lds[wave][1][0];   // X  / H1
  const int chBase = (lane < 16) ? 0 : 32;   // 2 lanes per query, 32 ch each

#pragma unroll 1
  for (int s = 0; s < 8; ++s) {
    Geo g = geom(cd, chc, cw, s);

    // -------- trilinear gather: 8 corners x 32 channels per lane ----------
    float facc[32];
#pragma unroll
    for (int i = 0; i < 32; ++i) facc[i] = 0.0f;
    int zz[2] = { g.z0, g.z1 }, yy[2] = { g.y0, g.y1 }, xx[2] = { g.x0, g.x1 };
#pragma unroll
    for (int c8 = 0; c8 < 8; ++c8) {
      int bz = (c8 >> 2) & 1, by = (c8 >> 1) & 1, bx = c8 & 1;
      float w = (bz ? g.wz : 1.0f - g.wz) * (by ? g.wy : 1.0f - g.wy) *
                (bx ? g.wx : 1.0f - g.wx);
      const v8h* p = (const v8h*)(vol2 +
          ((zz[bz] * 32 + yy[by]) * 32 + xx[bx]) * 64 + chBase);
#pragma unroll
      for (int j = 0; j < 4; ++j) {
        v8h d = p[j];
#pragma unroll
        for (int t = 0; t < 8; ++t)
          facc[j * 8 + t] = fmaf(w, (float)d[t], facc[j * 8 + t]);
      }
    }
    // -------- build X (16 x 96 f16) in LDS --------------------------------
    _Float16* xrow = bufB + r * KP0 + chBase;
#pragma unroll
    for (int j = 0; j < 4; ++j) {
      v8h o;
#pragma unroll
      for (int t = 0; t < 8; ++t) o[t] = (_Float16)facc[j * 8 + t];
      *(v8h*)(xrow + j * 8) = o;
    }
    if (lane < 16) {
      _Float16* trow = bufB + r * KP0 + 64;
      v8h t0 = {};
      t0[0] = (_Float16)g.rel0; t0[1] = (_Float16)g.rel1;
      t0[2] = (_Float16)g.rel2; t0[3] = (_Float16)rc0;
      t0[4] = (_Float16)rc1;    t0[5] = (_Float16)rc2;
      t0[6] = (_Float16)ssq;
      *(v8h*)trow = t0;
      v8h z8 = {};
      *(v8h*)(trow + 8)  = z8;
      *(v8h*)(trow + 16) = z8;
      *(v8h*)(trow + 24) = z8;
    }
    __syncthreads();

    // -------- 3 hidden layers via WMMA ------------------------------------
    mlp_layer<3>(bufB, KP0, bufA, W0h, b0, lane);   // X  -> H0
    __syncthreads();
    mlp_layer<8>(bufA, HID, bufB, W1h, b1, lane);   // H0 -> H1
    __syncthreads();
    mlp_layer<8>(bufB, HID, bufA, W2h, b2, lane);   // H1 -> H2
    __syncthreads();

    // -------- final 256 -> 1 dot (lanes 0..15, one row each) --------------
    if (lane < 16) {
      const _Float16* hp = bufA + r * HID;
      float dot = b3[0];
#pragma unroll 1
      for (int k = 0; k < 256; k += 8) {
        v8h h8 = *(const v8h*)(hp + k);
#pragma unroll
        for (int t = 0; t < 8; ++t) dot = fmaf((float)h8[t], W3[k + t], dot);
      }
      // areas list is reversed in the reference: pred[s] * area[7-s]
      outacc = fmaf(dot, geom(cd, chc, cw, 7 - s).area, outacc);
    }
    __syncthreads();
  }
  if (lane < 16) out[q] = outacc / tot;
}

// ---------------------------------------------------------------------------
// Host launch.  d_in order: inp, coord, cell, W0,b0, W1,b1, W2,b2, W3,b3.
// d_ws layout: [0,4MB) vol2 f16 | W0h 48KB | W1h 128KB | W2h 128KB.
// ---------------------------------------------------------------------------
extern "C" void kernel_launch(void* const* d_in, const int* in_sizes, int n_in,
                              void* d_out, int out_size, void* d_ws,
                              size_t ws_size, hipStream_t stream) {
  (void)in_sizes; (void)n_in; (void)out_size; (void)ws_size;
  const float* inp   = (const float*)d_in[0];
  const float* coord = (const float*)d_in[1];
  const float* cell  = (const float*)d_in[2];
  const float* W0 = (const float*)d_in[3];
  const float* b0 = (const float*)d_in[4];
  const float* W1 = (const float*)d_in[5];
  const float* b1 = (const float*)d_in[6];
  const float* W2 = (const float*)d_in[7];
  const float* b2 = (const float*)d_in[8];
  const float* W3 = (const float*)d_in[9];
  const float* b3 = (const float*)d_in[10];
  float* out = (float*)d_out;

  char* ws = (char*)d_ws;
  _Float16* vol2 = (_Float16*)(ws);                         // 4,194,304 B
  _Float16* W0h  = (_Float16*)(ws + 4u * 1024u * 1024u);    //    49,152 B
  _Float16* W1h  = (_Float16*)((char*)W0h + KP0 * HID * 2); //   131,072 B
  _Float16* W2h  = (_Float16*)((char*)W1h + HID * HID * 2); //   131,072 B

  liif_conv_vol<<<2097152 / 256, 256, 0, stream>>>(inp, vol2);
  liif_conv_w<<<(KP0 * HID + 2 * HID * HID) / 256, 256, 0, stream>>>(
      W0, W1, W2, W0h, W1h, W2h);
  liif_main<<<QTOT / 32, 64, 0, stream>>>(vol2, coord, cell,
                                          W0h, b0, W1h, b1, W2h, b2,
                                          W3, b3, out);
}
